// SplitAttention_61186104098913
// MI455X (gfx1250) — matrix operand order repaired and checked by reading, not compiled, not run on hardware
//
#include <hip/hip_runtime.h>
#include <hip/hip_bf16.h>

typedef __bf16 bf16;
typedef __attribute__((ext_vector_type(16))) __bf16 v16bf;
typedef __attribute__((ext_vector_type(8)))  __bf16 v8bf;
typedef __attribute__((ext_vector_type(8)))  float  v8f;

#define TOK_DIM  512
#define POS_DIM  512
#define N_HEADS  16
#define HEAD_DIM 64
#define D_MODEL  1024
#define INNER    1024
#define BB       4
#define TT       2048
#define MTOT     (BB*TT)   // 8192 tokens

// ---------------------------------------------------------------------------
// fragment helpers (ISA 7.12.2 layouts, wave32)
//   A 16x32 bf16: lane holds row m=lane%16; elems 0-7 -> K = 8*half + e,
//                 elems 8-15 -> K = 16 + 8*half + e
//   B 32x16 bf16: lane holds col n=lane%16; elems 0-7 -> K = 8*half + e,
//                 elems 8-15 -> K = 16 + 8*half + e
//   C/D 16x16 f32: lane holds col n=lane%16, vgpr r -> row m = r + 8*half
// ---------------------------------------------------------------------------
__device__ __forceinline__ v16bf cat16(v8bf lo, v8bf hi) {
  return __builtin_shufflevector(lo, hi, 0,1,2,3,4,5,6,7,8,9,10,11,12,13,14,15);
}
__device__ __forceinline__ v16bf load_frag_a(const bf16* p, int half) {
  v8bf lo = *reinterpret_cast<const v8bf*>(p + 8*half);
  v8bf hi = *reinterpret_cast<const v8bf*>(p + 16 + 8*half);
  return cat16(lo, hi);
}
__device__ __forceinline__ v16bf load_frag_b(const bf16* p, int half) {
  v8bf lo = *reinterpret_cast<const v8bf*>(p + 8*half);
  v8bf hi = *reinterpret_cast<const v8bf*>(p + 16 + 8*half);
  return cat16(lo, hi);
}
__device__ __forceinline__ v8f wmma_bf16(v16bf a, v16bf b, v8f c) {
  return __builtin_amdgcn_wmma_f32_16x16x32_bf16(false, a, false, b,
                                                 (short)0, c, false, false);
}

// ---------------------------------------------------------------------------
// Stage 0: fp32 -> bf16 converts
// ---------------------------------------------------------------------------
__global__ void split_convert_x(const float* __restrict__ x,
                                bf16* __restrict__ xtok,
                                bf16* __restrict__ xpos) {
  long idx = (long)blockIdx.x * blockDim.x + threadIdx.x;   // MTOT*1024 threads
  int row = (int)(idx >> 10);
  int col = (int)(idx & 1023);
  float v = x[idx];
  if (col < TOK_DIM) xtok[(long)row*TOK_DIM + col] = (bf16)v;
  else               xpos[(long)row*POS_DIM + (col - TOK_DIM)] = (bf16)v;
}

__global__ void convert_transpose(const float* __restrict__ W,
                                  bf16* __restrict__ Wt, int K, int N) {
  long idx = (long)blockIdx.x * blockDim.x + threadIdx.x;   // K*N threads
  int k = (int)(idx / N);
  int n = (int)(idx % N);
  Wt[(long)n*K + k] = (bf16)W[idx];
}

// ---------------------------------------------------------------------------
// Stage 1/3: bf16 WMMA GEMM.  C[M,1024] = A[M,K] @ Bt[1024,K]^T
//   wave tile 32x64 (2x4 accumulators), block = 8 waves -> 256x64
//   K compile-time -> fully unrolled ping-pong pipeline: literal load offsets,
//   no register rotation copies, loads overlap 8-WMMA groups.
//   MODE 0: bf16 row-major   MODE 1: bf16 per-head-transposed (V -> Vt)
//   MODE 2: fp32 row-major (final output)
// ---------------------------------------------------------------------------
template<int K, int MODE>
__global__ __launch_bounds__(256)
void gemm_bf16_wmma(const bf16* __restrict__ A, const bf16* __restrict__ Bt,
                    void* __restrict__ C, float scale) {
  constexpr int N = 1024;
  int lane = threadIdx.x & 31;
  int wave = __builtin_amdgcn_readfirstlane(threadIdx.x >> 5);  // uniform
  int half = lane >> 4;
  int lr   = lane & 15;
  int mbase = blockIdx.y * 256 + wave * 32;
  int nbase = blockIdx.x * 64;

  const bf16* Arow0 = A  + (long)(mbase + lr) * K;        // rows mbase+lr
  const bf16* Arow1 = Arow0 + (long)16 * K;               // rows mbase+16+lr
  const bf16* Brow0 = Bt + (long)(nbase + lr) * K;

  v8f acc[2][4] = {};

  // set X: fragments for current even k-step
  v16bf xa0 = load_frag_a(Arow0, half);
  v16bf xa1 = load_frag_a(Arow1, half);
  v16bf xb[4];
#pragma unroll
  for (int nt = 0; nt < 4; ++nt)
    xb[nt] = load_frag_b(Brow0 + nt*16*K, half);

#pragma unroll
  for (int kb = 0; kb < K; kb += 64) {
    // set Y: load kb+32 fragments (overlaps the 8 WMMAs on set X)
    v16bf ya0 = load_frag_a(Arow0 + kb + 32, half);
    v16bf ya1 = load_frag_a(Arow1 + kb + 32, half);
    v16bf yb[4];
#pragma unroll
    for (int nt = 0; nt < 4; ++nt)
      yb[nt] = load_frag_b(Brow0 + nt*16*K + kb + 32, half);

#pragma unroll
    for (int nt = 0; nt < 4; ++nt) {
      acc[0][nt] = wmma_bf16(xa0, xb[nt], acc[0][nt]);
      acc[1][nt] = wmma_bf16(xa1, xb[nt], acc[1][nt]);
    }

    // set X: load kb+64 fragments (overlaps the 8 WMMAs on set Y)
    if (kb + 64 < K) {
      xa0 = load_frag_a(Arow0 + kb + 64, half);
      xa1 = load_frag_a(Arow1 + kb + 64, half);
#pragma unroll
      for (int nt = 0; nt < 4; ++nt)
        xb[nt] = load_frag_b(Brow0 + nt*16*K + kb + 64, half);
    }

#pragma unroll
    for (int nt = 0; nt < 4; ++nt) {
      acc[0][nt] = wmma_bf16(ya0, yb[nt], acc[0][nt]);
      acc[1][nt] = wmma_bf16(ya1, yb[nt], acc[1][nt]);
    }
  }

#pragma unroll
  for (int mt = 0; mt < 2; ++mt) {
#pragma unroll
    for (int nt = 0; nt < 4; ++nt) {
#pragma unroll
      for (int r = 0; r < 8; ++r) {
        int m = mbase + mt*16 + r + 8*half;
        int n = nbase + nt*16 + lr;
        float v = acc[mt][nt][r] * scale;
        if (MODE == 0) {
          ((bf16*)C)[(long)m*N + n] = (bf16)v;
        } else if (MODE == 1) {            // V -> Vt[(b*H+h)*64+d][t]
          int b = m / TT, t = m % TT, h = n >> 6, d = n & 63;
          ((bf16*)C)[(long)((b*N_HEADS + h)*HEAD_DIM + d)*TT + t] = (bf16)v;
        } else {
          ((float*)C)[(long)m*N + n] = v;
        }
      }
    }
  }
}

// ---------------------------------------------------------------------------
// Stage 2: causal flash attention.  Wave = 16 queries, key blocks of 64.
//   Q,K in [b,t, h*64+d] bf16 (Q pre-scaled 1/8); Vt in [(b*H+h)*64+d, t].
//   Row sums of P computed on the matrix unit (P @ ones) instead of
//   ds_bpermute butterflies; only the row max uses lane shuffles.
// ---------------------------------------------------------------------------
__global__ __launch_bounds__(256)
void flash_attn_wmma(const bf16* __restrict__ Q, const bf16* __restrict__ Km,
                     const bf16* __restrict__ Vt, bf16* __restrict__ O) {
  __shared__ bf16 Plds[8][16][64];                 // 16 KB, per-wave slices
  int lane = threadIdx.x & 31;
  int wave = __builtin_amdgcn_readfirstlane(threadIdx.x >> 5);  // uniform
  int half = lane >> 4;
  int lr   = lane & 15;
  int bh = blockIdx.y;
  int b  = bh >> 4, h = bh & 15;
  int qbase = blockIdx.x * 128 + wave * 16;        // scalar -> scalar loop ctrl

  // constant all-ones B fragment for row-sum WMMA
  v16bf onesb;
#pragma unroll
  for (int i = 0; i < 16; ++i) onesb[i] = (bf16)1.0f;

  const bf16* Qrow = Q + ((long)(b*TT + qbase + lr) * INNER + h*HEAD_DIM);
  v16bf qf[2];
#pragma unroll
  for (int ks = 0; ks < 2; ++ks) qf[ks] = load_frag_a(Qrow + ks*32, half);

  v8f o[4] = {};
  float m_run[8], l_run[8];
#pragma unroll
  for (int r = 0; r < 8; ++r) { m_run[r] = -1e30f; l_run[r] = 0.0f; }

  const bf16* Kbase = Km + ((long)(b*TT + lr) * INNER + h*HEAD_DIM);
  const bf16* Vbase = Vt + ((long)(bh*HEAD_DIM + lr) * TT);

  for (int kbase = 0; kbase <= qbase + 15; kbase += 64) {
    // prefetch next key/value block behind this block's softmax VALU work
    if (kbase + 64 <= qbase + 15) {
      __builtin_prefetch(Kbase + (long)(kbase + 64) * INNER, 0, 3);
      __builtin_prefetch(Vbase + (kbase + 64), 0, 3);
    }

    // ---- S = (Q/sqrt(d)) @ K^T : 16 x 64 ----
    v8f s[4] = {};
    const bf16* Krow = Kbase + (long)kbase * INNER;
#pragma unroll
    for (int nt = 0; nt < 4; ++nt) {
      const bf16* Kr = Krow + (long)nt*16*INNER;
#pragma unroll
      for (int ks = 0; ks < 2; ++ks) {
        v16bf kf = load_frag_b(Kr + ks*32, half);
        s[nt] = wmma_bf16(qf[ks], kf, s[nt]);
      }
    }

    bool need_mask = (kbase + 63) > qbase;
    float alpha[8];
#pragma unroll
    for (int r = 0; r < 8; ++r) {
      int mq = qbase + r + 8*half;
      if (need_mask) {
#pragma unroll
        for (int nt = 0; nt < 4; ++nt) {
          int nk = kbase + nt*16 + lr;
          if (nk > mq) s[nt][r] = -1e30f;
        }
      }
      // row max across the 16 lanes of this half
      float mx = fmaxf(fmaxf(s[0][r], s[1][r]), fmaxf(s[2][r], s[3][r]));
#pragma unroll
      for (int off = 1; off < 16; off <<= 1)
        mx = fmaxf(mx, __shfl_xor(mx, off, 32));
      float mnew = fmaxf(m_run[r], mx);
      alpha[r] = __expf(m_run[r] - mnew);
      m_run[r] = mnew;

      // P = exp(S - mnew), D-layout -> LDS (row-major P tile)
#pragma unroll
      for (int nt = 0; nt < 4; ++nt)
        Plds[wave][r + 8*half][nt*16 + lr] = (bf16)__expf(s[nt][r] - mnew);
    }

    // same-wave LDS RAW: DS ops are in-order per wave; wait + compiler fence
    asm volatile("s_wait_dscnt 0" ::: "memory");

    v16bf pf[2];
#pragma unroll
    for (int ks = 0; ks < 2; ++ks)
      pf[ks] = load_frag_a(&Plds[wave][lr][ks*32], half);

    // row sums on the matrix unit: rsv[r] = sum_k P[row r+8*half][k]
    v8f rsv = {};
    rsv = wmma_bf16(pf[0], onesb, rsv);
    rsv = wmma_bf16(pf[1], onesb, rsv);
#pragma unroll
    for (int r = 0; r < 8; ++r)
      l_run[r] = l_run[r] * alpha[r] + rsv[r];

    // ---- O = O*alpha + P @ V ----
    const bf16* Vrow = Vbase + kbase;
#pragma unroll
    for (int dt = 0; dt < 4; ++dt) {
#pragma unroll
      for (int r = 0; r < 8; ++r) o[dt][r] *= alpha[r];
      const bf16* Vr = Vrow + (long)dt*16*TT;
#pragma unroll
      for (int ks = 0; ks < 2; ++ks) {
        v16bf vf = load_frag_b(Vr + ks*32, half);
        o[dt] = wmma_bf16(pf[ks], vf, o[dt]);
      }
    }
  }

  // epilogue: normalize and store bf16 attention output [b,t,h*64+d]
#pragma unroll
  for (int r = 0; r < 8; ++r) {
    float inv = 1.0f / l_run[r];
    int mq = qbase + r + 8*half;
#pragma unroll
    for (int dt = 0; dt < 4; ++dt)
      O[(long)(b*TT + mq) * INNER + h*HEAD_DIM + dt*16 + lr] =
          (bf16)(o[dt][r] * inv);
  }
}

// ---------------------------------------------------------------------------
extern "C" void kernel_launch(void* const* d_in, const int* in_sizes, int n_in,
                              void* d_out, int out_size, void* d_ws, size_t ws_size,
                              hipStream_t stream) {
  (void)in_sizes; (void)n_in; (void)out_size; (void)ws_size;
  const float* x  = (const float*)d_in[0];
  const float* Wq = (const float*)d_in[1];
  const float* Wk = (const float*)d_in[2];
  const float* Wv = (const float*)d_in[3];
  const float* Wo = (const float*)d_in[4];
  float* out = (float*)d_out;

  char* ws = (char*)d_ws;
  bf16* xtok = (bf16*)ws; ws += (size_t)MTOT*TOK_DIM*sizeof(bf16);
  bf16* xpos = (bf16*)ws; ws += (size_t)MTOT*POS_DIM*sizeof(bf16);
  bf16* WqT  = (bf16*)ws; ws += (size_t)INNER*POS_DIM*sizeof(bf16);
  bf16* WkT  = (bf16*)ws; ws += (size_t)INNER*POS_DIM*sizeof(bf16);
  bf16* WvT  = (bf16*)ws; ws += (size_t)INNER*TOK_DIM*sizeof(bf16);
  bf16* WoT  = (bf16*)ws; ws += (size_t)D_MODEL*INNER*sizeof(bf16);
  bf16* Qb   = (bf16*)ws; ws += (size_t)MTOT*INNER*sizeof(bf16);
  bf16* Kb   = (bf16*)ws; ws += (size_t)MTOT*INNER*sizeof(bf16);
  bf16* Vtb  = (bf16*)ws; ws += (size_t)MTOT*INNER*sizeof(bf16);
  bf16* AOut = (bf16*)ws; ws += (size_t)MTOT*INNER*sizeof(bf16);

  // converts
  split_convert_x<<<(MTOT*1024)/256, 256, 0, stream>>>(x, xtok, xpos);
  convert_transpose<<<(POS_DIM*INNER)/256, 256, 0, stream>>>(Wq, WqT, POS_DIM, INNER);
  convert_transpose<<<(POS_DIM*INNER)/256, 256, 0, stream>>>(Wk, WkT, POS_DIM, INNER);
  convert_transpose<<<(TOK_DIM*INNER)/256, 256, 0, stream>>>(Wv, WvT, TOK_DIM, INNER);
  convert_transpose<<<(INNER*D_MODEL)/256, 256, 0, stream>>>(Wo, WoT, INNER, D_MODEL);

  // projections (Q pre-scaled by 1/sqrt(HEAD_DIM))
  dim3 gg(INNER/64, MTOT/256);
  gemm_bf16_wmma<POS_DIM, 0><<<gg, 256, 0, stream>>>(xpos, WqT, Qb,  0.125f);
  gemm_bf16_wmma<POS_DIM, 0><<<gg, 256, 0, stream>>>(xpos, WkT, Kb,  1.0f);
  gemm_bf16_wmma<TOK_DIM, 1><<<gg, 256, 0, stream>>>(xtok, WvT, Vtb, 1.0f);

  // causal flash attention
  flash_attn_wmma<<<dim3(TT/128, BB*N_HEADS), 256, 0, stream>>>(Qb, Kb, Vtb, AOut);

  // output projection -> fp32
  gemm_bf16_wmma<INNER, 2><<<gg, 256, 0, stream>>>(AOut, WoT, out, 1.0f);
}
